// NonLocalLayer_10788957847969
// MI455X (gfx1250) — matrix operand order repaired and checked by reading, not compiled, not run on hardware
//
#include <hip/hip_runtime.h>
#include <math.h>

#define BSZ   4
#define CIN   1024
#define CI    256
#define COUT  1024
#define NSP   4096   // H*W = 64*64

typedef __attribute__((ext_vector_type(16))) __bf16        v16bf;
typedef __attribute__((ext_vector_type(8)))  float         v8f;
typedef __attribute__((ext_vector_type(8)))  unsigned int  v8u;

__device__ __forceinline__ unsigned short f2bf(float f) {
    union { float f; unsigned u; } a; a.f = f;
    unsigned r = a.u + 0x7fffu + ((a.u >> 16) & 1u);   // round-to-nearest-even
    return (unsigned short)(r >> 16);
}

__device__ __forceinline__ v16bf as_bf(v8u v) {
    union { v8u u; v16bf b; } c; c.u = v; return c.b;
}

// A operand: 16x32 bf16 tile, src row-major (M x K), ld in elements.
// lane 0-15: row=lane, K=0..7 (V0-3) and K=16..23 (V4-7)
// lane16-31: row=lane-16, K=8..15 and K=24..31
__device__ __forceinline__ v16bf load_a(const unsigned short* __restrict__ base, int ld) {
    int lane = threadIdx.x & 31;
    const unsigned short* p = base + (size_t)(lane & 15) * ld + ((lane >> 4) << 3);
    uint4 lo = *reinterpret_cast<const uint4*>(p);
    uint4 hi = *reinterpret_cast<const uint4*>(p + 16);
    v8u v = {lo.x, lo.y, lo.z, lo.w, hi.x, hi.y, hi.z, hi.w};
    return as_bf(v);
}

// B operand: 32x16 (K x N) bf16 tile, src is B^T row-major (N x K), ld in elements.
// lane 0-15: col n=lane, K=0..7 and K=16..23; lane 16-31: col n=lane-16, K=8..15 and K=24..31
__device__ __forceinline__ v16bf load_b(const unsigned short* __restrict__ base, int ld) {
    int lane = threadIdx.x & 31;
    const unsigned short* p = base + (size_t)(lane & 15) * ld + ((lane >> 4) << 3);
    uint4 lo = *reinterpret_cast<const uint4*>(p);
    uint4 hi = *reinterpret_cast<const uint4*>(p + 16);
    v8u v = {lo.x, lo.y, lo.z, lo.w, hi.x, hi.y, hi.z, hi.w};
    return as_bf(v);
}

__device__ __forceinline__ v8f wmma_bf16(v16bf a, v16bf b, v8f c) {
    return __builtin_amdgcn_wmma_f32_16x16x32_bf16(false, a, false, b, (short)0, c, false, false);
}

// ---------------------------------------------------------------- prep kernels
__global__ void k_cvt(const float* __restrict__ s, unsigned short* __restrict__ d, int n) {
    int i = blockIdx.x * blockDim.x + threadIdx.x;
    if (i < n) d[i] = f2bf(s[i]);
}

// x (B, CIN, NSP) f32  ->  xt (B, NSP, CIN) bf16
__global__ void k_transpose_x(const float* __restrict__ x, unsigned short* __restrict__ xt) {
    __shared__ float tile[32][33];
    int b  = blockIdx.z;
    int c0 = blockIdx.x * 32;
    int n0 = blockIdx.y * 32;
    const float* src = x + ((size_t)b * CIN + c0) * NSP + n0;
    for (int i = threadIdx.y; i < 32; i += 8)
        tile[i][threadIdx.x] = src[(size_t)i * NSP + threadIdx.x];   // tile[c][n]
    __syncthreads();
    unsigned short* dst = xt + ((size_t)b * NSP + n0) * CIN + c0;
    for (int i = threadIdx.y; i < 32; i += 8)
        dst[(size_t)i * CIN + threadIdx.x] = f2bf(tile[threadIdx.x][i]);
}

// ---------------------------------------------------------------- fused projection GEMM
// Shared B operand (xt rows) feeds BOTH weight matrices:
//   key:   qkT[b,n,c]  = BN(sum_k wk[c,k] xt[b,n,k] + b_key)      (B,NSP,CI) bf16
//   value: val[b,c,n]  =    sum_k wv[c,k] xt[b,n,k] + b_value     (B,CI,NSP) bf16
// Ping-pong double buffering (unroll-2) -> zero register-copy rotation.
__global__ void __launch_bounds__(128)
k_proj2(const unsigned short* __restrict__ wk, const unsigned short* __restrict__ wv,
        const unsigned short* __restrict__ xt,
        unsigned short* __restrict__ qkT, unsigned short* __restrict__ valout,
        const float* __restrict__ gamma, const float* __restrict__ beta,
        const float* __restrict__ mean,  const float* __restrict__ var,
        const float* __restrict__ bkey,  const float* __restrict__ bval)
{
    int wave = threadIdx.x >> 5, lane = threadIdx.x & 31;
    int c0 = blockIdx.x * 16;
    int n0 = blockIdx.y * 256 + wave * 64;
    int b  = blockIdx.z;
    const unsigned short* xb  = xt + (size_t)b * NSP * CIN;
    const unsigned short* wkr = wk + (size_t)c0 * CIN;
    const unsigned short* wvr = wv + (size_t)c0 * CIN;
    const unsigned short* xr[4];
#pragma unroll
    for (int t = 0; t < 4; ++t) xr[t] = xb + (size_t)(n0 + t * 16) * CIN;

    v8f acck[4] = {}, accv[4] = {};

    // buffer set 0: kk = 0
    v16bf ak0 = load_a(wkr, CIN), av0 = load_a(wvr, CIN);
    v16bf b0[4];
#pragma unroll
    for (int t = 0; t < 4; ++t) b0[t] = load_b(xr[t], CIN);

    for (int kk = 0; kk < CIN; kk += 64) {
        int k1 = kk + 32;
        // load buffer set 1 while consuming set 0
        v16bf ak1 = load_a(wkr + k1, CIN), av1 = load_a(wvr + k1, CIN);
        v16bf b1[4];
#pragma unroll
        for (int t = 0; t < 4; ++t) b1[t] = load_b(xr[t] + k1, CIN);
#pragma unroll
        for (int t = 0; t < 4; ++t) acck[t] = wmma_bf16(ak0, b0[t], acck[t]);
#pragma unroll
        for (int t = 0; t < 4; ++t) accv[t] = wmma_bf16(av0, b0[t], accv[t]);

        int k2 = (kk + 64 < CIN) ? kk + 64 : 0;   // last prefetch is discarded
        // reload buffer set 0 while consuming set 1
        ak0 = load_a(wkr + k2, CIN); av0 = load_a(wvr + k2, CIN);
#pragma unroll
        for (int t = 0; t < 4; ++t) b0[t] = load_b(xr[t] + k2, CIN);
#pragma unroll
        for (int t = 0; t < 4; ++t) acck[t] = wmma_bf16(ak1, b1[t], acck[t]);
#pragma unroll
        for (int t = 0; t < 4; ++t) accv[t] = wmma_bf16(av1, b1[t], accv[t]);
    }

    int hlf = lane >> 4, ncol = lane & 15;
#pragma unroll
    for (int r = 0; r < 8; ++r) {
        int c = c0 + r + hlf * 8;
        float inv    = gamma[c] * rsqrtf(var[c] + 1e-5f);
        float kshift = beta[c] + (bkey[c] - mean[c]) * inv;
        float vshift = bval[c];
#pragma unroll
        for (int t = 0; t < 4; ++t) {
            int n = n0 + t * 16 + ncol;
            qkT[((size_t)b * NSP + n) * CI + c]    = f2bf(acck[t][r] * inv + kshift);
            valout[((size_t)b * CI + c) * NSP + n] = f2bf(accv[t][r] + vshift);
        }
    }
}

// ---------------------------------------------------------------- flash attention
// qkT (B,NSP,CI) bf16 : query rows and key rows (same tensor)
// val (B,CI,NSP) bf16 : V^T row-major over channels
// ctx (B,NSP,CI) bf16 : softmax(Q K^T / 16) V
__global__ void __launch_bounds__(128)
k_attn(const unsigned short* __restrict__ qkT, const unsigned short* __restrict__ val,
       unsigned short* __restrict__ ctx)
{
    __shared__ __align__(16) unsigned short plds[4][16 * 32];
    int wave = threadIdx.x >> 5, lane = threadIdx.x & 31;
    int b  = blockIdx.y;
    int q0 = blockIdx.x * 64 + wave * 16;
    const unsigned short* qb = qkT + (size_t)b * NSP * CI;
    const unsigned short* vb = val + (size_t)b * CI * NSP;
    unsigned short* pl = plds[wave];
    int hlf = lane >> 4, ncol = lane & 15;

    v16bf qf[8];
#pragma unroll
    for (int f = 0; f < 8; ++f) qf[f] = load_a(qb + (size_t)q0 * CI + f * 32, CI);

    v8f cacc[16] = {};
    float rmax[8], rsum[8];
#pragma unroll
    for (int r = 0; r < 8; ++r) { rmax[r] = -1e30f; rsum[r] = 0.f; }

    for (int kb = 0; kb < NSP; kb += 32) {
        if (kb + 32 < NSP) __builtin_prefetch(qb + (size_t)(kb + 32) * CI, 0, 1);
        // ---- S = Q (16xCI) x K^T chunk (CI x 32): two 16x16 f32 tiles
        v8f s0 = {}, s1 = {};
#pragma unroll
        for (int f = 0; f < 8; ++f) {
            v16bf k0 = load_b(qb + (size_t)(kb +  0) * CI + f * 32, CI);
            v16bf k1 = load_b(qb + (size_t)(kb + 16) * CI + f * 32, CI);
            s0 = wmma_bf16(qf[f], k0, s0);
            s1 = wmma_bf16(qf[f], k1, s1);
        }
        // ---- issue all V fragment loads now; softmax VALU below hides them
        v16bf vf[16];
#pragma unroll
        for (int t = 0; t < 16; ++t)
            vf[t] = load_b(vb + (size_t)(t * 16) * NSP + kb, NSP);

        // ---- online softmax (rows live in (vgpr, lane-half); reduce over 16 lanes)
        const float sc = 0.0625f;                 // 256^-0.5
        float p0[8], p1[8], corr[8];
#pragma unroll
        for (int r = 0; r < 8; ++r) {
            float a0 = s0[r] * sc, a1 = s1[r] * sc;
            float m = fmaxf(a0, a1);
#pragma unroll
            for (int off = 1; off < 16; off <<= 1) m = fmaxf(m, __shfl_xor(m, off, 32));
            float nm = fmaxf(rmax[r], m);
            corr[r] = __expf(rmax[r] - nm);
            rmax[r] = nm;
            p0[r] = __expf(a0 - nm);
            p1[r] = __expf(a1 - nm);
            float sum = p0[r] + p1[r];
#pragma unroll
            for (int off = 1; off < 16; off <<= 1) sum += __shfl_xor(sum, off, 32);
            rsum[r] = rsum[r] * corr[r] + sum;
        }

        // ---- stage P (16x32 bf16) in LDS; rescale accumulators while DS is in flight
#pragma unroll
        for (int r = 0; r < 8; ++r) {
            int m = r + hlf * 8;
            pl[m * 32 + ncol]      = f2bf(p0[r]);
            pl[m * 32 + 16 + ncol] = f2bf(p1[r]);
        }
#pragma unroll
        for (int t = 0; t < 16; ++t)
#pragma unroll
            for (int r = 0; r < 8; ++r) cacc[t][r] *= corr[r];
        asm volatile("s_wait_dscnt 0x0" ::: "memory");
        v16bf pf = load_a(pl, 32);

        // ---- context += P (16x32) x V chunk (32 x CI): 16 independent WMMA chains
#pragma unroll
        for (int t = 0; t < 16; ++t)
            cacc[t] = wmma_bf16(pf, vf[t], cacc[t]);
        asm volatile("s_wait_dscnt 0x0" ::: "memory");   // pf loads done before next stores
    }

    unsigned short* cb = ctx + (size_t)b * NSP * CI;
#pragma unroll
    for (int r = 0; r < 8; ++r) {
        float inv = 1.f / rsum[r];
        int n = q0 + r + hlf * 8;
#pragma unroll
        for (int t = 0; t < 16; ++t)
            cb[(size_t)n * CI + t * 16 + ncol] = f2bf(cacc[t][r] * inv);
    }
}

// ---------------------------------------------------------------- output GEMM
// out[b,o,n] = sum_c w_out[o,c] * ctx[b,n,c] + b_out[o]   (fp32 output)
// 32x64 wave tile; K-loop (8 steps) fully unrolled -> scheduler hoists loads.
__global__ void __launch_bounds__(128)
k_out(const unsigned short* __restrict__ wbf, const unsigned short* __restrict__ ctx,
      const float* __restrict__ bout, float* __restrict__ out)
{
    int wave = threadIdx.x >> 5, lane = threadIdx.x & 31;
    int o0 = blockIdx.x * 32;
    int n0 = blockIdx.y * 256 + wave * 64;
    int b  = blockIdx.z;
    const unsigned short* cb  = ctx + (size_t)b * NSP * CI;
    const unsigned short* wr0 = wbf + (size_t)o0 * CI;
    const unsigned short* wr1 = wbf + (size_t)(o0 + 16) * CI;

    v8f acc[2][4] = {};
#pragma unroll
    for (int kk = 0; kk < CI; kk += 32) {
        v16bf a0 = load_a(wr0 + kk, CI);
        v16bf a1 = load_a(wr1 + kk, CI);
        v16bf bf[4];
#pragma unroll
        for (int t = 0; t < 4; ++t) bf[t] = load_b(cb + (size_t)(n0 + t * 16) * CI + kk, CI);
#pragma unroll
        for (int t = 0; t < 4; ++t) acc[0][t] = wmma_bf16(a0, bf[t], acc[0][t]);
#pragma unroll
        for (int t = 0; t < 4; ++t) acc[1][t] = wmma_bf16(a1, bf[t], acc[1][t]);
    }

    int hlf = lane >> 4, ncol = lane & 15;
#pragma unroll
    for (int m = 0; m < 2; ++m)
#pragma unroll
        for (int r = 0; r < 8; ++r) {
            int o = o0 + m * 16 + r + hlf * 8;
            float bo = bout[o];
#pragma unroll
            for (int t = 0; t < 4; ++t)
                out[((size_t)b * COUT + o) * NSP + n0 + t * 16 + ncol] = acc[m][t][r] + bo;
        }
}

// ---------------------------------------------------------------- host launch
extern "C" void kernel_launch(void* const* d_in, const int* in_sizes, int n_in,
                              void* d_out, int out_size, void* d_ws, size_t ws_size,
                              hipStream_t stream) {
    (void)in_sizes; (void)n_in; (void)out_size; (void)ws_size;
    const float* x       = (const float*)d_in[0];
    const float* w_key   = (const float*)d_in[1];
    const float* b_key   = (const float*)d_in[2];
    const float* gamma   = (const float*)d_in[3];
    const float* beta    = (const float*)d_in[4];
    const float* mean    = (const float*)d_in[5];
    const float* var     = (const float*)d_in[6];
    const float* w_value = (const float*)d_in[7];
    const float* b_value = (const float*)d_in[8];
    const float* w_out   = (const float*)d_in[9];
    const float* b_out   = (const float*)d_in[10];
    float* out = (float*)d_out;

    char* ws = (char*)d_ws;
    size_t off = 0;
    auto carve = [&](size_t bytes) {
        void* p = ws + off; off += (bytes + 255) & ~(size_t)255; return p;
    };
    unsigned short* xt   = (unsigned short*)carve((size_t)BSZ * NSP * CIN * 2);
    unsigned short* wkb  = (unsigned short*)carve((size_t)CI * CIN * 2);
    unsigned short* wvb  = (unsigned short*)carve((size_t)CI * CIN * 2);
    unsigned short* wob  = (unsigned short*)carve((size_t)COUT * CI * 2);
    unsigned short* qkT  = (unsigned short*)carve((size_t)BSZ * NSP * CI * 2);
    unsigned short* valb = (unsigned short*)carve((size_t)BSZ * CI * NSP * 2);
    unsigned short* ctx  = (unsigned short*)carve((size_t)BSZ * NSP * CI * 2);

    k_cvt<<<(CI * CIN + 255) / 256, 256, 0, stream>>>(w_key,   wkb, CI * CIN);
    k_cvt<<<(CI * CIN + 255) / 256, 256, 0, stream>>>(w_value, wvb, CI * CIN);
    k_cvt<<<(COUT * CI + 255) / 256, 256, 0, stream>>>(w_out,  wob, COUT * CI);

    k_transpose_x<<<dim3(CIN / 32, NSP / 32, BSZ), dim3(32, 8), 0, stream>>>(x, xt);

    k_proj2<<<dim3(CI / 16, NSP / 256, BSZ), 128, 0, stream>>>(
        wkb, wvb, xt, qkT, valb, gamma, beta, mean, var, b_key, b_value);

    k_attn<<<dim3(NSP / 64, BSZ), 128, 0, stream>>>(qkT, valb, ctx);

    k_out<<<dim3(COUT / 32, NSP / 256, BSZ), 128, 0, stream>>>(wob, ctx, b_out, out);
}
